// snn_mnist_44470091382848
// MI455X (gfx1250) — matrix operand order repaired and checked by reading, not compiled, not run on hardware
//
#include <hip/hip_runtime.h>

typedef __attribute__((ext_vector_type(16))) _Float16 v16h;
typedef __attribute__((ext_vector_type(8)))  float    v8f;

#define TSTEPS 100
#define BB 256
#define II 784
#define NN 1024
#define KCF 25            // forward K chunks of 32 (784 padded to 800)
#define KCS 8             // STDP K chunks of 32 (B=256)
#define NT (NN / 16)      // 64 neuron tiles
#define BT (BB / 16)      // 16 batch tiles
#define IT (II / 16)      // 49 input tiles

#define C_ALPHA   0.9f
#define C_BETA    0.8f
#define C_BPLUS   0.9f
#define C_BMINUS  0.9f
#define C_REF     5.0f
#define C_LAT     0.1f
#define C_BTHETA  0.99f
#define C_THADD   0.05f
#define C_LRB     (0.01f / 256.0f)

// All GEMM operands are pre-packed into per-wave fragment layout:
//   buf[tile][kchunk][lane][16 halves]  -> one 32B v16h load per lane (2x b128).
__device__ __forceinline__ v16h load_packed(const _Float16* __restrict__ tile,
                                            int kc, int lane) {
  return *(const v16h*)(tile + ((size_t)kc * 32 + lane) * 16);
}

// A-fragment offset (ISA 7.12.2 16-bit A layout) for element (row m, K index k):
// lane = m + ((k&8)?16:0) by low-16 K bit3; h = (k&7) + ((k&16)?8:0).
__device__ __forceinline__ size_t a_pack_off(int tileIdx, int kChunks, int k, int m) {
  const int kc = k >> 5, kk = k & 31, klo = kk & 15;
  const int lane = m + ((klo & 8) ? 16 : 0);
  const int h = (klo & 7) + ((kk & 16) ? 8 : 0);
  return ((size_t)(tileIdx * kChunks + kc) * 32 + lane) * 16 + h;
}

// B-fragment offset (same K order as A): lane = col + ((k&8)?16:0); h = (k&7) + ((k&16)?8:0).
__device__ __forceinline__ size_t b_pack_off(int tileIdx, int kChunks, int k, int col) {
  const int kc = k >> 5, kk = k & 31;
  const int lane = (col & 15) + ((kk & 8) ? 16 : 0);
  return ((size_t)(tileIdx * kChunks + kc) * 32 + lane) * 16 + (kk & 7) + ((kk & 16) ? 8 : 0);
}

__device__ __forceinline__ v8f wmma_f16(v16h a, v16h b, v8f c) {
  return __builtin_amdgcn_wmma_f32_16x16x32_f16(false, a, false, b, (short)0, c, false, false);
}

// ---------------- K0: per-call init --------------------------------------------------
__global__ void k0_init(const float* __restrict__ W, const float* __restrict__ x0,
                        float* __restrict__ Wf, _Float16* __restrict__ WhTp,
                        _Float16* __restrict__ xAp,
                        float* __restrict__ syn, float* __restrict__ mem,
                        float* __restrict__ spk, float* __restrict__ theta,
                        float* __restrict__ ref, float* __restrict__ post,
                        float* __restrict__ preTr, unsigned* __restrict__ cnt) {
  int idx = blockIdx.x * blockDim.x + threadIdx.x;
  int stride = gridDim.x * blockDim.x;
  // copy W master + pack into forward-B fragments [NT][KCF][32][16]
  for (int p = idx; p < NN * II; p += stride) {
    float w = W[p];
    Wf[p] = w;
    int n = p / II, i = p - n * II;
    WhTp[b_pack_off(n >> 4, KCF, i, n)] = (_Float16)w;
  }
  // WhTp K-tail (i=784..799 -> h>=8 slots of kc=24): zero once; never rewritten.
  for (int p = idx; p < NT * 16 * 16; p += stride) {
    int nt = p >> 8, r = p & 255;
    WhTp[((size_t)(nt * KCF + (KCF - 1)) * 32 + (r >> 3)) * 16 + 8 + (r & 7)] =
        (_Float16)0.0f;
  }
  // stage x for step 0 as forward-A fragments [BT][KCF][32][16]
  for (int p = idx; p < BB * II; p += stride) {
    int b = p / II, i = p - b * II;
    xAp[a_pack_off(b >> 4, KCF, i, b & 15)] = (_Float16)x0[p];
  }
  // xAp K-tail (i=784..799 -> h>=8 slots of kc=24): zero once; staging keeps h<8 there.
  for (int p = idx; p < BT * 32 * 8; p += stride) {
    int bt = p >> 8, r = p & 255;
    xAp[((size_t)(bt * KCF + (KCF - 1)) * 32 + (r >> 3)) * 16 + 8 + (r & 7)] =
        (_Float16)0.0f;
  }
  for (int p = idx; p < BB * NN; p += stride) {
    syn[p] = 0.f; mem[p] = 0.f; spk[p] = 0.f;
    theta[p] = 0.f; ref[p] = 0.f; post[p] = 0.f;
  }
  for (int p = idx; p < BB * II; p += stride) preTr[p] = 0.f;
  for (int p = idx; p < TSTEPS; p += stride) cnt[p] = 0u;
}

// ---------------- K2: forward WMMA GEMM + fused LIF/refractory epilogue --------------
__global__ __launch_bounds__(256) void k2_forward(const _Float16* __restrict__ xAp,
                                                  const _Float16* __restrict__ WhTp,
                                                  float* __restrict__ syn,
                                                  float* __restrict__ mem,
                                                  float* __restrict__ spk,
                                                  const float* __restrict__ theta,
                                                  float* __restrict__ ref,
                                                  float* __restrict__ out_t,
                                                  unsigned* __restrict__ cnt_t) {
  const int lane = threadIdx.x & 31;
  const int wave = threadIdx.x >> 5;
  const int bt = blockIdx.y;                       // batch tile
  const int b0 = bt * 16;
  const int nt = blockIdx.x * 8 + wave;            // neuron tile (one per wave)
  const int n0 = nt * 16;

  const _Float16* atile = xAp  + (size_t)bt * KCF * 512;
  const _Float16* btile = WhTp + (size_t)nt * KCF * 512;

  v8f acc = {};
#pragma unroll 5
  for (int kc = 0; kc < KCF; ++kc) {               // padded K: tails are exact zeros
    acc = wmma_f16(load_packed(atile, kc, lane), load_packed(btile, kc, lane), acc);
  }

  // D layout: lane&15 -> n column (coalesced), r + 8*(lane>=16) -> b row.
  const int n = n0 + (lane & 15);
  const int mrow = (lane & 16) ? 8 : 0;
  int local_spikes = 0;
#pragma unroll
  for (int r = 0; r < 8; ++r) {
    int b = b0 + mrow + r;
    int idx = b * NN + n;
    float pre   = acc[r];
    float th    = 1.0f + theta[idx];
    float sp    = spk[idx];                        // previous step's spike
    float rt    = ref[idx] + sp * C_REF;
    float s_old = syn[idx], m_old = mem[idx];
    float s_c   = C_ALPHA * s_old + pre;
    float m_c   = C_BETA * m_old + s_c - sp * th;  // reset-by-subtract
    float spn   = (m_c > th) ? 1.0f : 0.0f;
    if (rt > 0.0f) { spn = 0.0f; m_c = m_old; s_c = s_old; }
    syn[idx] = s_c;
    mem[idx] = m_c;
    spk[idx] = spn;
    ref[idx] = rt - 1.0f;
    out_t[idx] = m_c;                              // mem_rec[t]
    local_spikes += (spn > 0.0f) ? 1 : 0;
  }

  __shared__ int red[256];
  red[threadIdx.x] = local_spikes;
  __syncthreads();
  for (int s = 128; s > 0; s >>= 1) {
    if (threadIdx.x < s) red[threadIdx.x] += red[threadIdx.x + s];
    __syncthreads();
  }
  if (threadIdx.x == 0 && red[0] > 0) atomicAdd(cnt_t, (unsigned)red[0]);
}

// -- K3: argmax + inhibition + theta/post/pre traces + operand staging (incl. x_{t+1}) -
__global__ __launch_bounds__(256) void k3_inhibit(const float* __restrict__ x,
                                                  const float* __restrict__ xnext,
                                                  float* __restrict__ syn,
                                                  const float* __restrict__ mem,
                                                  const float* __restrict__ spk,
                                                  float* __restrict__ theta,
                                                  float* __restrict__ post,
                                                  float* __restrict__ preTr,
                                                  _Float16* __restrict__ spkAp,
                                                  _Float16* __restrict__ postAp,
                                                  _Float16* __restrict__ xBp,
                                                  _Float16* __restrict__ preTrBp,
                                                  _Float16* __restrict__ xAp,
                                                  const unsigned* __restrict__ cnt_t) {
  const int b = blockIdx.x;                        // one block per sample
  const int tid = threadIdx.x;
  __shared__ float vmax[256];
  __shared__ int   vidx[256];

  float best = -3.0e38f; int besti = 0;
  for (int n = tid; n < NN; n += 256) {            // increasing n -> first-max kept
    float v = mem[b * NN + n];
    if (v > best) { best = v; besti = n; }
  }
  vmax[tid] = best; vidx[tid] = besti;
  __syncthreads();
  for (int s = 128; s > 0; s >>= 1) {
    if (tid < s) {
      float v2 = vmax[tid + s]; int i2 = vidx[tid + s];
      if (v2 > vmax[tid] || (v2 == vmax[tid] && i2 < vidx[tid])) {
        vmax[tid] = v2; vidx[tid] = i2;
      }
    }
    __syncthreads();
  }
  const int winner = vidx[0];
  const bool any = (*cnt_t) > 0u;

  for (int n = tid; n < NN; n += 256) {
    int idx = b * NN + n;
    float s = syn[idx];
    if (any) {
      float m = (n == winner) ? 0.0f : 1.0f;
      s = fmaxf(s - m * C_LAT, 0.0f);              // clip applies to winner too
    }
    syn[idx] = s;
    float sp = spk[idx];
    theta[idx] = C_BTHETA * theta[idx] + C_THADD * sp;
    float po = C_BMINUS * post[idx] + sp;
    post[idx] = po;
    // STDP A operands, fragment-packed: M = n, K = b
    spkAp[a_pack_off(n >> 4, KCS, b, n & 15)]  = (_Float16)sp;
    postAp[a_pack_off(n >> 4, KCS, b, n & 15)] = (_Float16)po;
  }

  // pre-trace decay + STDP B staging (this step's x_t; only k4 consumes pre_trace)
  for (int i = tid; i < II; i += 256) {
    int idx = b * II + i;
    float xv = x[idx];
    float pt = C_BPLUS * preTr[idx] + xv;
    preTr[idx] = pt;
    size_t off = b_pack_off(i >> 4, KCS, b, i);
    xBp[off]     = (_Float16)xv;
    preTrBp[off] = (_Float16)pt;
  }

  // pre-stage next step's x as forward-A fragments (k2(t) already consumed xAp)
  if (xnext) {
    for (int i = tid; i < II; i += 256) {
      xAp[a_pack_off(b >> 4, KCF, i, b & 15)] = (_Float16)xnext[b * II + i];
    }
  }
}

// ---------------- K4: STDP dual-WMMA GEMM + clamped W update -------------------------
__global__ __launch_bounds__(256) void k4_stdp(const _Float16* __restrict__ spkAp,
                                               const _Float16* __restrict__ postAp,
                                               const _Float16* __restrict__ preTrBp,
                                               const _Float16* __restrict__ xBp,
                                               float* __restrict__ Wf,
                                               _Float16* __restrict__ WhTp) {
  const int lane = threadIdx.x & 31;
  const int wave = threadIdx.x >> 5;
  const int it = blockIdx.x;                       // input tile (49)
  const int i0 = it * 16;
  const int nt = blockIdx.y * 8 + wave;            // neuron tile
  const int n0 = nt * 16;

  const _Float16* a1tile = spkAp   + (size_t)nt * KCS * 512;
  const _Float16* a2tile = postAp  + (size_t)nt * KCS * 512;
  const _Float16* b1tile = preTrBp + (size_t)it * KCS * 512;
  const _Float16* b2tile = xBp     + (size_t)it * KCS * 512;

  v8f acc1 = {}, acc2 = {};
#pragma unroll
  for (int kc = 0; kc < KCS; ++kc) {               // K = 256, no tail
    acc1 = wmma_f16(load_packed(a1tile, kc, lane), load_packed(b1tile, kc, lane), acc1);
    acc2 = wmma_f16(load_packed(a2tile, kc, lane), load_packed(b2tile, kc, lane), acc2);
  }

  const int i = i0 + (lane & 15);
  const int mrow = (lane & 16) ? 8 : 0;
#pragma unroll
  for (int r = 0; r < 8; ++r) {
    int n = n0 + mrow + r;
    float w = Wf[(size_t)n * II + i] + C_LRB * (acc1[r] - acc2[r]);
    w = fminf(fmaxf(w, 0.0f), 1.0f);
    Wf[(size_t)n * II + i] = w;                    // fp32 master (coalesced)
    WhTp[b_pack_off(nt, KCF, i, n)] = (_Float16)w; // packed forward-B for next step
  }
}

// ---------------- launch -------------------------------------------------------------
extern "C" void kernel_launch(void* const* d_in, const int* in_sizes, int n_in,
                              void* d_out, int out_size, void* d_ws, size_t ws_size,
                              hipStream_t stream) {
  (void)in_sizes; (void)n_in; (void)out_size; (void)ws_size;
  const float* image = (const float*)d_in[0];      // [T, B, I] fp32 (0/1 spikes)
  const float* W     = (const float*)d_in[1];      // [N, I] fp32 (const; copied to ws)
  float* out = (float*)d_out;                      // [T, B, N] fp32

  char* ws = (char*)d_ws;
  size_t off = 0;
  auto carve = [&](size_t bytes) -> void* {
    void* p = ws + off;
    off = (off + bytes + 255) & ~(size_t)255;
    return p;
  };
  float*    Wf      = (float*)carve((size_t)NN * II * 4);
  float*    syn     = (float*)carve((size_t)BB * NN * 4);
  float*    mem     = (float*)carve((size_t)BB * NN * 4);
  float*    spk     = (float*)carve((size_t)BB * NN * 4);
  float*    theta   = (float*)carve((size_t)BB * NN * 4);
  float*    ref     = (float*)carve((size_t)BB * NN * 4);
  float*    post    = (float*)carve((size_t)BB * NN * 4);
  float*    preTr   = (float*)carve((size_t)BB * II * 4);
  unsigned* cnt     = (unsigned*)carve((size_t)TSTEPS * 4);
  _Float16* WhTp    = (_Float16*)carve((size_t)NT * KCF * 512 * 2);  // forward B (packed)
  _Float16* xAp     = (_Float16*)carve((size_t)BT * KCF * 512 * 2);  // forward A (packed)
  _Float16* xBp     = (_Float16*)carve((size_t)IT * KCS * 512 * 2);  // STDP B (packed)
  _Float16* preTrBp = (_Float16*)carve((size_t)IT * KCS * 512 * 2);  // STDP B (packed)
  _Float16* spkAp   = (_Float16*)carve((size_t)NT * KCS * 512 * 2);  // STDP A (packed)
  _Float16* postAp  = (_Float16*)carve((size_t)NT * KCS * 512 * 2);  // STDP A (packed)

  k0_init<<<512, 256, 0, stream>>>(W, image, Wf, WhTp, xAp, syn, mem, spk, theta, ref,
                                   post, preTr, cnt);

  for (int t = 0; t < TSTEPS; ++t) {
    const float* xt = image + (size_t)t * BB * II;
    const float* xn = (t + 1 < TSTEPS) ? (image + (size_t)(t + 1) * BB * II) : nullptr;
    k2_forward<<<dim3(8, 16), 256, 0, stream>>>(xAp, WhTp, syn, mem, spk, theta, ref,
                                                out + (size_t)t * BB * NN, cnt + t);
    k3_inhibit<<<BB, 256, 0, stream>>>(xt, xn, syn, mem, spk, theta, post, preTr,
                                       spkAp, postAp, xBp, preTrBp, xAp, cnt + t);
    k4_stdp<<<dim3(49, 8), 256, 0, stream>>>(spkAp, postAp, preTrBp, xBp, Wf, WhTp);
  }
}
